// StructureLayer_74002286510096
// MI455X (gfx1250) — matrix operand order repaired and checked by reading, not compiled, not run on hardware
//
#include <hip/hip_runtime.h>

// ---------------------------------------------------------------------------
// CDNA5 (gfx1250) wave32 WMMA types
// ---------------------------------------------------------------------------
typedef __attribute__((ext_vector_type(16))) __bf16 v16bf;
typedef __attribute__((ext_vector_type(8)))  float  v8f;

__device__ __forceinline__ __bf16 bf(float x) { return (__bf16)x; }

// ---------------------------------------------------------------------------
// Generic bf16-WMMA GEMM: C[M,N] = A[M,K] @ B[K,N] (+bias) (opt. ReLU)
// One wave per 16x64 output tile: 4 accumulators, A fragment loaded once per
// K-step and reused across 4 back-to-back WMMAs. B panel (32x64) staged in
// LDS: coalesced float4 loads when a 16-col subtile is fully inside N,
// branchless clamped-address loads otherwise. Subtile count nsub is
// block-uniform so EXEC stays all-ones around the WMMAs.
// bf16 16x16x32 operand layouts as used here (A/B contiguous k; C per ISA):
//  A (16x32): lane=half*16+q holds row m=q; elem e -> k = 16*half + e
//  B (32x16): lane holds col n=q;           elem e -> k = 16*half + e
//  C (f32)  : lane holds col n=q; vgpr r -> row m = r + 8*half
// ---------------------------------------------------------------------------
__global__ void k_gemm(const float* __restrict__ A, const float* __restrict__ B,
                       const float* __restrict__ bias, float* __restrict__ C,
                       int N, int K, int lda, int ldb, int ldc,
                       long long sA, long long sB, long long sC, int relu)
{
  __shared__ float lsB[2048];                      // 32k x 64n f32 panel
  const float* Ab = A + (size_t)blockIdx.z * sA;
  const float* Bb = B + (size_t)blockIdx.z * sB;
  float*       Cb = C + (size_t)blockIdx.z * sC;
  int tm = blockIdx.y << 4, tn = blockIdx.x << 6;
  int lane = threadIdx.x & 31;
  int half = lane >> 4, ql = lane & 15;
  int m = tm + ql;
  int nsub = (N - tn + 15) >> 4; nsub = nsub > 4 ? 4 : nsub;
  bool ldb4 = (ldb & 3) == 0;
  const float* Arow = Ab + (size_t)m * lda;

  v8f acc[4] = {};
  for (int k0 = 0; k0 < K; k0 += 32) {
#pragma unroll
    for (int sub = 0; sub < 4; ++sub) {
      if (sub >= nsub) break;                      // block-uniform
      int tc = tn + (sub << 4);
      float* ls = lsB + (sub << 9);
      if (ldb4 && (tc + 16 <= N)) {
        int row = lane >> 2;                       // 0..7
        int col = (lane & 3) << 2;                 // 0,4,8,12
#pragma unroll
        for (int r2 = 0; r2 < 4; ++r2) {
          int rr = row + (r2 << 3);
          float4 v = *(const float4*)(Bb + (size_t)(k0 + rr) * ldb + tc + col);
          *(float4*)(&ls[(rr << 4) + col]) = v;
        }
      } else {
        const float* Brow = Bb + (size_t)(k0 + lane) * ldb;
#pragma unroll
        for (int c = 0; c < 16; ++c) {
          int cg = tc + c;
          int cc = cg < N ? cg : 0;                // clamp: load always valid
          float v = Brow[cc];
          ls[(lane << 4) + c] = (cg < N) ? v : 0.f;
        }
      }
    }
    __syncthreads();
    v16bf a;
#pragma unroll
    for (int e = 0; e < 16; ++e) {
      int ka = k0 + e + (half << 4);
      a[e] = bf(Arow[ka]);
    }
#pragma unroll
    for (int sub = 0; sub < 4; ++sub) {
      if (sub >= nsub) break;
      v16bf b;
#pragma unroll
      for (int e = 0; e < 16; ++e)
        b[e] = bf(lsB[(sub << 9) + (((half << 4) + e) << 4) + ql]);
      acc[sub] = __builtin_amdgcn_wmma_f32_16x16x32_bf16(false, a, false, b,
                                                         (short)0, acc[sub],
                                                         false, false);
    }
    __syncthreads();
  }
#pragma unroll
  for (int sub = 0; sub < 4; ++sub) {
    if (sub >= nsub) break;
    int n = tn + (sub << 4) + ql;
    if (n < N) {
      float bb = bias ? bias[n] : 0.f;
#pragma unroll
      for (int r = 0; r < 8; ++r) {
        int mo = tm + r + (half << 3);
        float v = acc[sub][r] + bb;
        if (relu) v = fmaxf(v, 0.f);
        Cb[(size_t)mo * ldc + n] = v;
      }
    }
  }
}

// ---------------------------------------------------------------------------
// Rotate/translate projected points into global frame.
// grid 512, block 192 (48 qp points + 144 kvp points per residue)
// ---------------------------------------------------------------------------
__global__ void k_prep(const float* __restrict__ qpr, const float* __restrict__ kvpr,
                       const float* __restrict__ rot, const float* __restrict__ trans,
                       float* __restrict__ qpo, float* __restrict__ kp,
                       float* __restrict__ vpf)
{
  int n = blockIdx.x, t = threadIdx.x;
  const float* R = rot + n * 9;
  float t0 = trans[n * 3 + 0] * 0.1f;
  float t1 = trans[n * 3 + 1] * 0.1f;
  float t2 = trans[n * 3 + 2] * 0.1f;
  if (t < 48) {
    float px = qpr[n * 144 + t], py = qpr[n * 144 + 48 + t], pz = qpr[n * 144 + 96 + t];
    float ox = R[0] * px + R[1] * py + R[2] * pz + t0;
    float oy = R[3] * px + R[4] * py + R[5] * pz + t1;
    float oz = R[6] * px + R[7] * py + R[8] * pz + t2;
    int h = t >> 2, p = t & 3;
    float* d = qpo + n * 144 + h * 12 + p * 3;
    d[0] = ox; d[1] = oy; d[2] = oz;
  } else if (t < 192) {
    int j = t - 48;
    float px = kvpr[n * 432 + j], py = kvpr[n * 432 + 144 + j], pz = kvpr[n * 432 + 288 + j];
    float ox = R[0] * px + R[1] * py + R[2] * pz + t0;
    float oy = R[3] * px + R[4] * py + R[5] * pz + t1;
    float oz = R[6] * px + R[7] * py + R[8] * pz + t2;
    int h = j / 12, p = j % 12;
    if (p < 4) {
      float* d = kp + n * 144 + h * 12 + p * 3;
      d[0] = ox; d[1] = oy; d[2] = oz;
    } else {
      float* d = vpf + n * 288 + h * 24 + (p - 4) * 3;
      d[0] = ox; d[1] = oy; d[2] = oz;
    }
  }
}

// Sq[n*12+h] = |qp[n,h,:12]|^2 ; Sk likewise from kp.
__global__ void k_sums(const float* __restrict__ qpo, const float* __restrict__ kp,
                       float* __restrict__ Sq, float* __restrict__ Sk)
{
  int idx = blockIdx.x * 128 + threadIdx.x;
  if (idx >= 512 * 12) return;
  int n = idx / 12, h = idx % 12;
  const float* a = qpo + n * 144 + h * 12;
  const float* b = kp  + n * 144 + h * 12;
  float s1 = 0.f, s2 = 0.f;
#pragma unroll
  for (int d = 0; d < 12; ++d) { s1 += a[d] * a[d]; s2 += b[d] * b[d]; }
  Sq[idx] = s1; Sk[idx] = s2;
}

// ---------------------------------------------------------------------------
// Pack the fused-logits WMMA operands into contiguous 32-float K rows:
//  qk_a[n,h,:32] = [ q*1/sqrt(48) (16) | qp*hw (12) | 0 0 0 0 ]
//  qk_b[n,h,:32] = [ k           (16) | kp    (12) | 0 0 0 0 ]
// grid 512, block 384 (t = h*32+d)
// ---------------------------------------------------------------------------
__global__ void k_pack(const float* __restrict__ q, const float* __restrict__ kv,
                       const float* __restrict__ qpo, const float* __restrict__ kp,
                       const float* __restrict__ head_w,
                       float* __restrict__ qk_a, float* __restrict__ qk_b)
{
  int n = blockIdx.x, t = threadIdx.x;
  int h = t >> 5, d = t & 31;
  float x = head_w[h];
  float sp = (x > 20.f) ? x : log1pf(expf(x));      // softplus
  float hw = sp * 0.13608276348795434f;             // * 1/sqrt(54)
  float va = 0.f, vb = 0.f;
  if (d < 16) {
    va = q[n * 192 + h * 16 + d] * 0.14433756729740643f;  // 1/sqrt(48)
    vb = kv[n * 384 + h * 32 + d];
  } else if (d < 28) {
    va = qpo[n * 144 + h * 12 + (d - 16)] * hw;
    vb = kp[n * 144 + h * 12 + (d - 16)];
  }
  qk_a[n * 384 + t] = va;
  qk_b[n * 384 + t] = vb;
}

// ---------------------------------------------------------------------------
// Fused attention logits: one block per (i-tile, head) computes the full
// 16x512 strip; packed A fragment, softplus, Sq and mask rows hoisted and
// reused across 32 back-to-back WMMAs. Only B rows stream per j-tile
// (contiguous 64B per lane).
//   logits = wmma + sqrt(1/3)*zb - 0.5*hw*(Sq+Sk) + (mask_i*mask_j-1)*1e9
// grid (32 it, 12 h), block 32 (one wave)
// ---------------------------------------------------------------------------
__global__ void k_logits(const float* __restrict__ qk_a, const float* __restrict__ qk_b,
                         const float* __restrict__ Sq, const float* __restrict__ Sk,
                         const float* __restrict__ zb, const float* __restrict__ mask,
                         const float* __restrict__ head_w, float* __restrict__ logits)
{
  int h = blockIdx.y;
  int tm = blockIdx.x << 4;
  int lane = threadIdx.x & 31, half = lane >> 4, ql = lane & 15;
  int m = tm + ql;
  const float* arow = qk_a + m * 384 + h * 32;

  v16bf av;
#pragma unroll
  for (int e = 0; e < 16; ++e) {
    int kk = e + (half << 4);
    av[e] = bf(arow[kk]);
  }
  float x = head_w[h];
  float sp = (x > 20.f) ? x : log1pf(expf(x));
  float hw = sp * 0.13608276348795434f;
  float sqr[8], mr[8];
#pragma unroll
  for (int r = 0; r < 8; ++r) {
    int mi = tm + r + (half << 3);
    sqr[r] = Sq[mi * 12 + h];
    mr[r]  = mask[mi];
  }

  for (int tn = 0; tn < 512; tn += 16) {
    int n = tn + ql;
    const float* brow = qk_b + n * 384 + h * 32 + (half << 4);
    v16bf bv;
#pragma unroll
    for (int e = 0; e < 16; ++e) bv[e] = bf(brow[e]);
    v8f acc = {};
    acc = __builtin_amdgcn_wmma_f32_16x16x32_bf16(false, av, false, bv,
                                                  (short)0, acc, false, false);
    float mn  = mask[n];
    float skn = Sk[n * 12 + h];
#pragma unroll
    for (int r = 0; r < 8; ++r) {
      int mi = tm + r + (half << 3);
      float val = acc[r]
                + 0.5773502691896258f * zb[((size_t)mi * 512 + n) * 12 + h]
                - 0.5f * hw * (sqr[r] + skn)
                + (mr[r] * mn - 1.f) * 1e9f;
      logits[((size_t)h * 512 + mi) * 512 + n] = val;
    }
  }
}

// Row softmax over j, in place. grid (512 i, 12 h), block 32 (one wave).
__global__ void k_softmax(float* __restrict__ logits)
{
  int i = blockIdx.x, h = blockIdx.y;
  float* row = logits + ((size_t)h * 512 + i) * 512;
  int lane = threadIdx.x;
  float mx = -3.4e38f;
  for (int j = lane; j < 512; j += 32) mx = fmaxf(mx, row[j]);
  for (int o = 16; o; o >>= 1) mx = fmaxf(mx, __shfl_xor(mx, o, 32));
  float sum = 0.f;
  for (int j = lane; j < 512; j += 32) {
    float e = expf(row[j] - mx);
    row[j] = e; sum += e;
  }
  for (int o = 16; o; o >>= 1) sum += __shfl_xor(sum, o, 32);
  float inv = 1.f / sum;
  for (int j = lane; j < 512; j += 32) row[j] *= inv;
}

// ---------------------------------------------------------------------------
// opair[i,h,:] = sum_j a[h,i,j] * z[i,j,:]  -> per-i WMMA (12x512)@(512x128).
// 16x64 tile per wave: A (attention rows, head-clamped, no guarded loads)
// gathered once per K-step and reused across 4 WMMAs against the LDS-staged
// z panel. Writes feat cols 576..2111. grid (2, 512), block 32.
// ---------------------------------------------------------------------------
__global__ void k_opair(const float* __restrict__ a, const float* __restrict__ z,
                        float* __restrict__ feat)
{
  __shared__ float lsB[2048];                      // 32 x 64 panel of z
  int i = blockIdx.y;
  int tn = blockIdx.x << 6;                        // 0 or 64
  int lane = threadIdx.x & 31, half = lane >> 4, ql = lane & 15;
  int arow = (ql < 12) ? ql : 0;                   // clamp to a valid row
  float amul = (ql < 12) ? 1.f : 0.f;              // zero padded rows by select
  const float* abase = a + ((size_t)arow * 512 + i) * 512;
  const float* zbase = z + (size_t)i * 512 * 128 + tn;

  v8f acc[4] = {};
  for (int k0 = 0; k0 < 512; k0 += 32) {
    int row = lane >> 2;
    int col = (lane & 3) << 2;
#pragma unroll
    for (int sub = 0; sub < 4; ++sub) {
#pragma unroll
      for (int r2 = 0; r2 < 4; ++r2) {
        int rr = row + (r2 << 3);
        float4 v = *(const float4*)(zbase + (size_t)(k0 + rr) * 128 + (sub << 4) + col);
        *(float4*)(&lsB[(sub << 9) + (rr << 4) + col]) = v;
      }
    }
    __syncthreads();
    v16bf av;
#pragma unroll
    for (int e = 0; e < 16; ++e) {
      int kk = k0 + e + (half << 4);
      av[e] = bf(abase[kk] * amul);
    }
#pragma unroll
    for (int sub = 0; sub < 4; ++sub) {
      v16bf bv;
#pragma unroll
      for (int e = 0; e < 16; ++e)
        bv[e] = bf(lsB[(sub << 9) + (((half << 4) + e) << 4) + ql]);
      acc[sub] = __builtin_amdgcn_wmma_f32_16x16x32_bf16(false, av, false, bv,
                                                         (short)0, acc[sub],
                                                         false, false);
    }
    __syncthreads();
  }
#pragma unroll
  for (int sub = 0; sub < 4; ++sub) {
    int n = tn + (sub << 4) + ql;
#pragma unroll
    for (int r = 0; r < 8; ++r) {
      int h = r + (half << 3);
      if (h < 12) feat[(size_t)i * 2112 + 576 + h * 128 + n] = acc[sub][r];
    }
  }
}

// Inverse-frame rotate attended points, write x/y/z/norm sections of feat.
// grid 512, block 96 (t = h*8+p).
__global__ void k_optpost(const float* __restrict__ optr, const float* __restrict__ rot,
                          const float* __restrict__ trans, float* __restrict__ feat)
{
  int n = blockIdx.x, t = threadIdx.x;
  int h = t >> 3, p = t & 7;
  const float* v = optr + n * 288 + h * 24 + p * 3;
  float dx = v[0] - trans[n * 3 + 0] * 0.1f;
  float dy = v[1] - trans[n * 3 + 1] * 0.1f;
  float dz = v[2] - trans[n * 3 + 2] * 0.1f;
  const float* R = rot + n * 9;
  float ox = R[0] * dx + R[3] * dy + R[6] * dz;   // R^T (einsum 'bnyx')
  float oy = R[1] * dx + R[4] * dy + R[7] * dz;
  float oz = R[2] * dx + R[5] * dy + R[8] * dz;
  size_t base = (size_t)n * 2112;
  feat[base + 192 + t] = ox;
  feat[base + 288 + t] = oy;
  feat[base + 384 + t] = oz;
  feat[base + 480 + t] = sqrtf(ox * ox + oy * oy + oz * oz + 1e-8f);
}

// out = LayerNorm(x + y) * g + b   (row width 384). grid 512, block 384.
__global__ void k_addln(const float* __restrict__ x, const float* __restrict__ y,
                        const float* __restrict__ g, const float* __restrict__ b,
                        float* __restrict__ outp)
{
  __shared__ float sh[384];
  __shared__ float stat;
  int i = blockIdx.x, t = threadIdx.x;
  float v = x[(size_t)i * 384 + t] + y[(size_t)i * 384 + t];
  sh[t] = v; __syncthreads();
  for (int s = 192; s >= 3; s >>= 1) { if (t < s) sh[t] += sh[t + s]; __syncthreads(); }
  if (t == 0) stat = (sh[0] + sh[1] + sh[2]) * (1.f / 384.f);
  __syncthreads();
  float mean = stat;
  float d = v - mean;
  __syncthreads();
  sh[t] = d * d; __syncthreads();
  for (int s = 192; s >= 3; s >>= 1) { if (t < s) sh[t] += sh[t + s]; __syncthreads(); }
  if (t == 0) stat = (sh[0] + sh[1] + sh[2]) * (1.f / 384.f);
  __syncthreads();
  float rstd = rsqrtf(stat + 1e-5f);
  outp[(size_t)i * 384 + t] = d * rstd * g[t] + b[t];
}

// Quaternion update of rigid frames; writes rot_new/trans_new into d_out.
__global__ void k_final(const float* __restrict__ u, const float* __restrict__ rot,
                        const float* __restrict__ trans, float* __restrict__ outp)
{
  int n = blockIdx.x * 64 + threadIdx.x;
  if (n >= 512) return;
  float a = u[n * 6 + 0], b2 = u[n * 6 + 1], c = u[n * 6 + 2];
  float inv = rsqrtf(1.f + a * a + b2 * b2 + c * c);
  float w = inv, x = a * inv, y = b2 * inv, zc = c * inv;
  float Ru[9];
  Ru[0] = 1.f - 2.f * (y * y + zc * zc); Ru[1] = 2.f * (x * y - w * zc);  Ru[2] = 2.f * (x * zc + w * y);
  Ru[3] = 2.f * (x * y + w * zc);        Ru[4] = 1.f - 2.f * (x * x + zc * zc); Ru[5] = 2.f * (y * zc - w * x);
  Ru[6] = 2.f * (x * zc - w * y);        Ru[7] = 2.f * (y * zc + w * x);  Ru[8] = 1.f - 2.f * (x * x + y * y);
  const float* R = rot + n * 9;
  float* ro = outp + 196608 + n * 9;
#pragma unroll
  for (int i2 = 0; i2 < 3; ++i2)
#pragma unroll
    for (int j = 0; j < 3; ++j)
      ro[i2 * 3 + j] = R[i2 * 3 + 0] * Ru[j] + R[i2 * 3 + 1] * Ru[3 + j] + R[i2 * 3 + 2] * Ru[6 + j];
  float tx = u[n * 6 + 3] * 10.f, ty = u[n * 6 + 4] * 10.f, tz = u[n * 6 + 5] * 10.f;
  float* to = outp + 201216 + n * 3;
  to[0] = R[0] * tx + R[1] * ty + R[2] * tz + trans[n * 3 + 0];
  to[1] = R[3] * tx + R[4] * ty + R[5] * tz + trans[n * 3 + 1];
  to[2] = R[6] * tx + R[7] * ty + R[8] * tz + trans[n * 3 + 2];
}

// ---------------------------------------------------------------------------
extern "C" void kernel_launch(void* const* d_in, const int* in_sizes, int n_in,
                              void* d_out, int out_size, void* d_ws, size_t ws_size,
                              hipStream_t stream)
{
  (void)in_sizes; (void)n_in; (void)out_size; (void)ws_size;
  const float* s     = (const float*)d_in[0];
  const float* z     = (const float*)d_in[1];
  const float* rot   = (const float*)d_in[2];
  const float* trans = (const float*)d_in[3];
  const float* mask  = (const float*)d_in[4];
  const float* Wq    = (const float*)d_in[5];
  const float* bq    = (const float*)d_in[6];
  const float* Wkv   = (const float*)d_in[7];
  const float* bkv   = (const float*)d_in[8];
  const float* Wqp   = (const float*)d_in[9];
  const float* bqp   = (const float*)d_in[10];
  const float* Wkvp  = (const float*)d_in[11];
  const float* bkvp  = (const float*)d_in[12];
  const float* Wb    = (const float*)d_in[13];
  const float* bbp   = (const float*)d_in[14];
  const float* headw = (const float*)d_in[15];
  const float* Wout  = (const float*)d_in[16];
  const float* bout  = (const float*)d_in[17];
  const float* ln1g  = (const float*)d_in[18];
  const float* ln1b  = (const float*)d_in[19];
  const float* Wt1   = (const float*)d_in[20];
  const float* bt1   = (const float*)d_in[21];
  const float* Wt2   = (const float*)d_in[22];
  const float* bt2   = (const float*)d_in[23];
  const float* Wt3   = (const float*)d_in[24];
  const float* bt3   = (const float*)d_in[25];
  const float* lntg  = (const float*)d_in[26];
  const float* lntb  = (const float*)d_in[27];
  const float* Wu    = (const float*)d_in[28];
  const float* bu    = (const float*)d_in[29];
  float* out = (float*)d_out;
  float* ws  = (float*)d_ws;

  // workspace layout (float offsets)
  float* q_    = ws;             // 512*192
  float* kv_   = ws + 98304;     // 512*384
  float* qpr_  = ws + 294912;    // 512*144
  float* kvpr_ = ws + 368640;    // 512*432
  float* qpo_  = ws + 589824;    // 512*144
  float* kp_   = ws + 663552;    // 512*144
  float* vpf_  = ws + 737280;    // 512*288
  float* Sq_   = ws + 884736;    // 512*12
  float* Sk_   = ws + 890880;    // 512*12
  float* zb_   = ws + 897024;    // 512*512*12
  float* lg_   = ws + 4042752;   // 12*512*512
  float* feat_ = ws + 7188480;   // 512*2112
  float* optr_ = ws + 8269824;   // 512*288
  float* out1_ = ws + 8417280;   // 512*384
  float* s1_   = ws + 8613888;   // 512*384
  float* t1_   = ws + 8810496;   // 512*384
  float* t2_   = ws + 9007104;   // 512*384
  float* t3_   = ws + 9203712;   // 512*384
  float* u_    = ws + 9400320;   // 512*6
  float* qka_  = ws + 9403392;   // 512*384 packed logits A
  float* qkb_  = ws + 9600000;   // 512*384 packed logits B

  dim3 wv(32, 1, 1);

  // 1) projections of s (bf16 WMMA GEMMs, 16x64 tiles)
  k_gemm<<<dim3(3, 32, 1), wv, 0, stream>>>(s, Wq,   bq,   q_,    192, 384, 384, 192, 192, 0LL, 0LL, 0LL, 0);
  k_gemm<<<dim3(6, 32, 1), wv, 0, stream>>>(s, Wkv,  bkv,  kv_,   384, 384, 384, 384, 384, 0LL, 0LL, 0LL, 0);
  k_gemm<<<dim3(3, 32, 1), wv, 0, stream>>>(s, Wqp,  bqp,  qpr_,  144, 384, 384, 144, 144, 0LL, 0LL, 0LL, 0);
  k_gemm<<<dim3(7, 32, 1), wv, 0, stream>>>(s, Wkvp, bkvp, kvpr_, 432, 384, 384, 432, 432, 0LL, 0LL, 0LL, 0);

  // 2) pair bias: zb = z @ Wb + bb  (M = 512*512 rows, K=128, N=12)
  k_gemm<<<dim3(1, 16384, 1), wv, 0, stream>>>(z, Wb, bbp, zb_, 12, 128, 128, 12, 12, 0LL, 0LL, 0LL, 0);

  // 3) rotate points into global frame; squared norms; pack logits operands
  k_prep<<<dim3(512, 1, 1), dim3(192, 1, 1), 0, stream>>>(qpr_, kvpr_, rot, trans, qpo_, kp_, vpf_);
  k_sums<<<dim3(48, 1, 1), dim3(128, 1, 1), 0, stream>>>(qpo_, kp_, Sq_, Sk_);
  k_pack<<<dim3(512, 1, 1), dim3(384, 1, 1), 0, stream>>>(q_, kv_, qpo_, kp_, headw, qka_, qkb_);

  // 4) fused logits (QK + point term in one WMMA, 16x512 strip/block) + softmax
  k_logits<<<dim3(32, 12, 1), wv, 0, stream>>>(qka_, qkb_, Sq_, Sk_, zb_, mask, headw, lg_);
  k_softmax<<<dim3(512, 12, 1), wv, 0, stream>>>(lg_);

  // 5) attention outputs (batched over heads via blockIdx.z)
  //    o = a @ v  -> feat cols [h*16 .. h*16+15]
  k_gemm<<<dim3(1, 32, 12), wv, 0, stream>>>(lg_, kv_ + 16, nullptr, feat_,
                                             16, 512, 512, 384, 2112,
                                             262144LL, 32LL, 16LL, 0);
  //    opt_raw = a @ vp (24 cols per head)
  k_gemm<<<dim3(1, 32, 12), wv, 0, stream>>>(lg_, vpf_, nullptr, optr_,
                                             24, 512, 512, 288, 288,
                                             262144LL, 24LL, 24LL, 0);
  //    opair = a_i @ z_i  (per residue i)
  k_opair<<<dim3(2, 512, 1), wv, 0, stream>>>(lg_, z, feat_);

  // 6) inverse rotate attended points; x/y/z/norm sections of feat
  k_optpost<<<dim3(512, 1, 1), dim3(96, 1, 1), 0, stream>>>(optr_, rot, trans, feat_);

  // 7) output projection + residual LN
  k_gemm<<<dim3(6, 32, 1), wv, 0, stream>>>(feat_, Wout, bout, out1_, 384, 2112, 2112, 384, 384, 0LL, 0LL, 0LL, 0);
  k_addln<<<dim3(512, 1, 1), dim3(384, 1, 1), 0, stream>>>(s, out1_, ln1g, ln1b, s1_);

  // 8) transition MLP + residual LN (s2 written straight into d_out)
  k_gemm<<<dim3(6, 32, 1), wv, 0, stream>>>(s1_, Wt1, bt1, t1_, 384, 384, 384, 384, 384, 0LL, 0LL, 0LL, 1);
  k_gemm<<<dim3(6, 32, 1), wv, 0, stream>>>(t1_, Wt2, bt2, t2_, 384, 384, 384, 384, 384, 0LL, 0LL, 0LL, 1);
  k_gemm<<<dim3(6, 32, 1), wv, 0, stream>>>(t2_, Wt3, bt3, t3_, 384, 384, 384, 384, 384, 0LL, 0LL, 0LL, 0);
  k_addln<<<dim3(512, 1, 1), dim3(384, 1, 1), 0, stream>>>(s1_, t3_, lntg, lntb, out);

  // 9) frame update: u = s2 @ Wu + bu ; quaternion compose
  k_gemm<<<dim3(1, 32, 1), wv, 0, stream>>>(out, Wu, bu, u_, 6, 384, 384, 6, 6, 0LL, 0LL, 0LL, 0);
  k_final<<<dim3(8, 1, 1), dim3(64, 1, 1), 0, stream>>>(u_, rot, trans, out);
}
